// CausalSelfAttention_75720273428698
// MI455X (gfx1250) — matrix operand order repaired and checked by reading, not compiled, not run on hardware
//
#include <hip/hip_runtime.h>
#include <stdint.h>
#include <math.h>

// ---------------------------------------------------------------------------
// CDNA5 (gfx1250) causal self-attention, bf16 WMMA end-to-end.
// B=2, T=2048, C=1024, H=16, D=64.
//   x --cvt--> bf16 --GEMM1(wmma)--> Q(pre-scaled by 0.125*log2e), K, V^T
//   flash attention: TDM double-buffered K/V tiles in LDS, base-2 online
//                    softmax, wmma for K Q^T and V^T P^T
//   GEMM3(wmma) --> fp32 out
// ---------------------------------------------------------------------------

typedef __attribute__((ext_vector_type(16))) __bf16 v16bf;
typedef __attribute__((ext_vector_type(8)))  float v8f;
typedef __attribute__((ext_vector_type(4)))  unsigned int v4u;
typedef __attribute__((ext_vector_type(8)))  int v8i;
typedef __attribute__((ext_vector_type(4)))  int v4i;
typedef unsigned short u16;
typedef unsigned int   u32;
typedef unsigned long long u64;

#ifndef __has_builtin
#define __has_builtin(x) 0
#endif
#if __has_builtin(__builtin_amdgcn_tensor_load_to_lds)
#define HAVE_TDM 1
#else
#define HAVE_TDM 0
#endif

#define WMMA_BF16(a, b, c) \
  __builtin_amdgcn_wmma_f32_16x16x32_bf16(false, (a), false, (b), (short)0, (c), false, false)

// 1/sqrt(64) * log2(e): folds the softmax scale and the base-2 exp into Q.
#define Q_SCALE 0.18033688f

__device__ __forceinline__ u16 f2bf(float f) {
  union { float f; u32 u; } x; x.f = f;
  u32 u = x.u;
  u += 0x7fffu + ((u >> 16) & 1u);   // round to nearest even
  return (u16)(u >> 16);
}

union FragU { v16bf v; uint4 q[2]; u16 us[16]; };

// A-fragment (16x32, M=row, K dim): lane holds row (lane&15); elements 0..15 are
// K = k0 + (lane>>4)*16 .. +15 (contiguous).
__device__ __forceinline__ v16bf load_frag_a(const u16* base, int ld, int row, int k0, int lane) {
  const u16* p = base + (size_t)row * ld + k0 + (lane >> 4) * 16;
  FragU u;
  u.q[0] = *(const uint4*)(p);
  u.q[1] = *(const uint4*)(p + 8);
  return u.v;
}

// B-fragment (32x16, K x N): lane holds column; 16 contiguous K values at
// k0 + (lane>>4)*16.  "row" here is the row of a row-major [N,K] matrix.
__device__ __forceinline__ v16bf load_frag_b(const u16* base, int ld, int row, int k0, int lane) {
  const u16* p = base + (size_t)row * ld + k0 + (lane >> 4) * 16;
  FragU u;
  u.q[0] = *(const uint4*)(p);
  u.q[1] = *(const uint4*)(p + 8);
  return u.v;
}

#if HAVE_TDM
// Issue one 2-D TDM tile load: tile_d1 rows of tile_d0 bf16 elements, row
// stride stride0 (elements), into contiguous LDS at lds_byte_addr.
// D# packing per CDNA5 ISA §8.3/§8.4 (group0 128b, group1 256b, 2-D tile so
// groups 2/3 are zero/disabled).
__device__ __forceinline__ void tdm_load_2d(u32 lds_byte_addr, const void* gptr,
                                            u32 tensor_d0, u32 tensor_d1,
                                            u32 tile_d0, u32 tile_d1, u32 stride0) {
  const u64 ga = (u64)(uintptr_t)gptr;                 // byte address
  v4u g0;
  g0.x = 1u;                                           // count=1 (valid user D#)
  g0.y = lds_byte_addr;                                // lds_addr [63:32]
  g0.z = (u32)ga;                                      // global_addr [95:64]
  g0.w = (u32)((ga >> 32) & 0x1FFFFFFu) | (2u << 30);  // addr[56:32], type=2
  v8i g1;
  g1[0] = (int)(1u << 16);                             // data_size=1 -> 2 bytes
  g1[1] = (int)(tensor_d0 << 16);                      // tensor_dim0[15:0]
  g1[2] = (int)((tensor_d0 >> 16) | (tensor_d1 << 16));// dim0[31:16]|dim1[15:0]
  g1[3] = (int)(((tensor_d1 >> 16) & 0xFFFFu) | (tile_d0 << 16)); // dim1 hi | tile_dim0
  g1[4] = (int)(tile_d1 & 0xFFFFu);                    // tile_dim1, tile_dim2=0
  g1[5] = (int)stride0;                                // tensor_dim0_stride lo32
  g1[6] = 0;                                           // stride0 hi, dim1_stride lo
  g1[7] = 0;                                           // dim1_stride hi
  v4i gz = {};
#if __clang_major__ >= 23
  v8i gz8 = {};
  __builtin_amdgcn_tensor_load_to_lds(g0, g1, gz, gz, gz8, 0);
#else
  __builtin_amdgcn_tensor_load_to_lds(g0, g1, gz, gz, 0);
#endif
}

// Generic pointer to a __shared__ object: low 32 bits are the LDS byte offset
// (LDS aperture truncation, ISA §10.2).
__device__ __forceinline__ u32 lds_off(const void* p) { return (u32)(uintptr_t)p; }
#endif

// ---------------------------------------------------------------------------
// fp32 -> bf16 conversion
// ---------------------------------------------------------------------------
__global__ __launch_bounds__(256) void cvt_f32_bf16(const float* __restrict__ in,
                                                    u16* __restrict__ out, int n) {
  int i = blockIdx.x * 256 + threadIdx.x;
  if (i < n) out[i] = f2bf(in[i]);
}

// ---------------------------------------------------------------------------
// GEMM1: qkv = x @ w_attn.T   (M=4096, N=3072, K=1024), scatter epilogue:
//   q -> Qb[b,h,t,d] * Q_SCALE   (row-major per head)
//   k -> Kb[b,h,t,d]
//   v -> Vt[b,h,d,t]             (transposed per head)
// One wave computes a 64x64 tile (4x4 WMMA accumulators).
// ---------------------------------------------------------------------------
__global__ __launch_bounds__(128) void gemm_qkv_kernel(const u16* __restrict__ X,
                                                       const u16* __restrict__ W,
                                                       u16* __restrict__ Qb,
                                                       u16* __restrict__ Kb,
                                                       u16* __restrict__ Vt) {
  const int lane = threadIdx.x & 31;
  const int gw   = blockIdx.x * 4 + (threadIdx.x >> 5);   // 3072 waves
  const int m0   = (gw / 48) * 64;
  const int n0   = (gw % 48) * 64;
  const int r    = lane & 15;
  const int g    = lane >> 4;

  v8f acc[4][4] = {};
  for (int k0 = 0; k0 < 1024; k0 += 32) {
    v16bf a[4], b[4];
#pragma unroll
    for (int i = 0; i < 4; ++i) a[i] = load_frag_a(X, 1024, m0 + i * 16 + r, k0, lane);
#pragma unroll
    for (int j = 0; j < 4; ++j) b[j] = load_frag_b(W, 1024, n0 + j * 16 + r, k0, lane);
    if (k0 + 32 < 1024) {   // speculative L2 prefetch of next K-slab
      __builtin_prefetch(&X[(size_t)(m0 + r) * 1024 + k0 + 32], 0, 3);
      __builtin_prefetch(&W[(size_t)(n0 + r) * 1024 + k0 + 32], 0, 3);
    }
#pragma unroll
    for (int i = 0; i < 4; ++i)
#pragma unroll
      for (int j = 0; j < 4; ++j)
        acc[i][j] = WMMA_BF16(a[i], b[j], acc[i][j]);
  }

#pragma unroll
  for (int i = 0; i < 4; ++i) {
#pragma unroll
    for (int j = 0; j < 4; ++j) {
      const int n = n0 + j * 16 + r;
      const int which = n >> 10;          // 0=q 1=k 2=v
      const int c = n & 1023;
      const int h = c >> 6, d = c & 63;
      const int mb = m0 + i * 16 + 8 * g;
#pragma unroll
      for (int v = 0; v < 8; ++v) {
        const int m  = mb + v;
        const int bb = m >> 11;           // batch
        const int t  = m & 2047;          // time
        const size_t hd = (size_t)(bb * 16 + h);
        const float val = acc[i][j][v];
        if (which == 0)      Qb[(hd * 2048 + t) * 64 + d] = f2bf(val * Q_SCALE);
        else if (which == 1) Kb[(hd * 2048 + t) * 64 + d] = f2bf(val);
        else                 Vt[(hd * 64 + d) * 2048 + t] = f2bf(val);
      }
    }
  }
}

// ---------------------------------------------------------------------------
// Flash attention (causal, base-2 online softmax).
// Block = 4 waves on one (b,h); wave w owns the 16-query tile (qt_base + w).
// Each 32-key chunk of K (32x64) and V^T (64x32) lives in LDS, double
// buffered: wave 0 issues TENSOR_LOAD_TO_LDS for chunk i+1 while all waves
// compute on chunk i (TENSORcnt completes in order, so waiting <=2 means the
// current pair is resident).  S^T = K Q^T so the softmax reduction is 16
// in-lane elements + one half-wave __shfl_xor; then O^T = V^T P^T.
// ---------------------------------------------------------------------------
__global__ __launch_bounds__(128) void flash_attn_kernel(const u16* __restrict__ Qb,
                                                         const u16* __restrict__ Kb,
                                                         const u16* __restrict__ Vt,
                                                         u16* __restrict__ Y) {
  __shared__ u16 sK[2][32 * 64];   // keys kc..kc+31, row-major [32][64]
  __shared__ u16 sV[2][64 * 32];   // V^T rows d=0..63, cols kc..kc+31, [64][32]

  const int tid  = threadIdx.x;
  const int lane = tid & 31;
  const int wid  = tid >> 5;
  const int bh      = blockIdx.x >> 5;            // 0..31  (b*16+h)
  const int qt_base = (blockIdx.x & 31) * 4;      // first q-tile of this block
  const int q0   = (qt_base + wid) * 16;          // this wave's query base
  const int g    = lane >> 4;
  const int qc   = lane & 15;
  const int q    = q0 + qc;                       // this lane's query

  const u16* Qh = Qb + (size_t)bh * 2048 * 64;
  const u16* Kh = Kb + (size_t)bh * 2048 * 64;
  const u16* Vh = Vt + (size_t)bh * 64 * 2048;

  // Q^T as B-fragments (d-chunks 0 and 32); register-resident for the loop.
  const v16bf qf0 = load_frag_b(Qh, 64, q0 + qc, 0, lane);
  const v16bf qf1 = load_frag_b(Qh, 64, q0 + qc, 32, lane);

  v8f o[4] = {};
  float mrun = -__builtin_inff();
  float lrun = 0.f;

  const int kend_wave = q0 + 16;                  // this wave's causal bound
  const int kend_blk  = qt_base * 16 + 64;        // last wave's causal bound

#if HAVE_TDM
  if (wid == 0) {   // prime the pipeline: chunk 0 into buffer 0
    tdm_load_2d(lds_off(&sK[0][0]), Kh,  64, 2048, 64, 32, 64);
    tdm_load_2d(lds_off(&sV[0][0]), Vh, 2048, 64, 32, 64, 2048);
  }
#endif

  int it = 0;
  for (int kc = 0; kc < kend_blk; kc += 32, ++it) {
    const int cur = it & 1;

#if HAVE_TDM
    if (wid == 0) {
      if (kc + 32 < kend_blk) {   // prefetch next chunk into the other buffer
        const int nb = cur ^ 1;
        tdm_load_2d(lds_off(&sK[nb][0]), Kh + (size_t)(kc + 32) * 64, 64, 2048, 64, 32, 64);
        tdm_load_2d(lds_off(&sV[nb][0]), Vh + (kc + 32), 2048, 64, 32, 64, 2048);
        __builtin_amdgcn_s_wait_tensorcnt(2);   // current pair resident
      } else {
        __builtin_amdgcn_s_wait_tensorcnt(0);
      }
    }
#else
    { // cooperative staging fallback: 128 threads move 4KB of K and 4KB of V^T
      const int idx = tid * 16;                   // u16 index, 32B per thread
      const int kr = idx >> 6, kcol = idx & 63;   // K: [32][64]
      const uint4* ks = (const uint4*)&Kh[(size_t)(kc + kr) * 64 + kcol];
      uint4 k0v = ks[0], k1v = ks[1];
      const int vr = idx >> 5, vcol = idx & 31;   // V^T: [64][32]
      const uint4* vs = (const uint4*)&Vh[(size_t)vr * 2048 + kc + vcol];
      uint4 v0v = vs[0], v1v = vs[1];
      *(uint4*)&sK[cur][idx] = k0v;  *(uint4*)&sK[cur][idx + 8] = k1v;
      *(uint4*)&sV[cur][idx] = v0v;  *(uint4*)&sV[cur][idx + 8] = v1v;
    }
#endif
    __syncthreads();

    if (kc < kend_wave) {   // wave-uniform: EXEC stays all-ones inside
      // S^T tiles: tile0 = keys kc..kc+15, tile1 = kc+16..kc+31 (16k x 16q)
      v8f s0 = {}, s1 = {};
      {
        v16bf kf;
        kf = load_frag_a(sK[cur], 64, qc,      0,  lane); s0 = WMMA_BF16(kf, qf0, s0);
        kf = load_frag_a(sK[cur], 64, qc,      32, lane); s0 = WMMA_BF16(kf, qf1, s0);
        kf = load_frag_a(sK[cur], 64, 16 + qc, 0,  lane); s1 = WMMA_BF16(kf, qf0, s1);
        kf = load_frag_a(sK[cur], 64, 16 + qc, 32, lane); s1 = WMMA_BF16(kf, qf1, s1);
      }

      // Causal mask (skipped entirely for chunks fully below the diagonal).
      // Lane owns column q; its elements are keys kc+8g+v / kc+16+8g+v.
      float sv[16];
      if (kc + 31 <= q0) {
#pragma unroll
        for (int v = 0; v < 8; ++v) { sv[v] = s0[v]; sv[8 + v] = s1[v]; }
      } else {
#pragma unroll
        for (int v = 0; v < 8; ++v) {
          sv[v]     = (kc + 8 * g + v      <= q) ? s0[v] : -__builtin_inff();
          sv[8 + v] = (kc + 16 + 8 * g + v <= q) ? s1[v] : -__builtin_inff();
        }
      }

      float mloc = sv[0];
#pragma unroll
      for (int i = 1; i < 16; ++i) mloc = fmaxf(mloc, sv[i]);
      mloc = fmaxf(mloc, __shfl_xor(mloc, 16, 32));
      const float mnew  = fmaxf(mrun, mloc);
      const float alpha = __builtin_exp2f(mrun - mnew);   // native v_exp_f32

      float p[16];
      float psum = 0.f;
#pragma unroll
      for (int i = 0; i < 16; ++i) { p[i] = __builtin_exp2f(sv[i] - mnew); psum += p[i]; }
      psum += __shfl_xor(psum, 16, 32);
      lrun = lrun * alpha + psum;
      mrun = mnew;

#pragma unroll
      for (int dt = 0; dt < 4; ++dt)
#pragma unroll
        for (int v = 0; v < 8; ++v) o[dt][v] *= alpha;

      // Re-layout P^T from C-frag to B-frag: element i of lane (g,qc) must
      // hold key kc + g*16 + i.  One 8-float half-wave exchange suffices.
      FragU pf;
#pragma unroll
      for (int v = 0; v < 8; ++v) {
        const float send = g ? p[v] : p[8 + v];     // what the partner needs
        const float recv = __shfl_xor(send, 16, 32);
        const float lo = g ? recv : p[v];
        const float hi = g ? p[8 + v] : recv;
        pf.us[v]     = f2bf(lo);
        pf.us[8 + v] = f2bf(hi);
      }

      // O^T += V^T(d-tile, k-chunk) * P^T
#pragma unroll
      for (int dt = 0; dt < 4; ++dt) {
        const v16bf vf = load_frag_a(sV[cur], 32, dt * 16 + qc, 0, lane);
        o[dt] = WMMA_BF16(vf, pf.v, o[dt]);
      }
    }
    __syncthreads();   // all waves done with this buffer
  }

  // Epilogue: O[q,d] = acc/l, packed bf16 row-major into Y[B*T, C].
  const float linv = 1.f / lrun;
  const int bb = bh >> 4, hh = bh & 15;
  const size_t mrow = (size_t)bb * 2048 + q;
#pragma unroll
  for (int dt = 0; dt < 4; ++dt) {
    union { uint4 q4; u16 us[8]; } st;
#pragma unroll
    for (int v = 0; v < 8; ++v) st.us[v] = f2bf(o[dt][v] * linv);
    *(uint4*)&Y[mrow * 1024 + hh * 64 + dt * 16 + 8 * g] = st.q4;
  }
}

// ---------------------------------------------------------------------------
// GEMM3: out = y @ w_proj.T   (M=4096, N=1024, K=1024), fp32 output.
// ---------------------------------------------------------------------------
__global__ __launch_bounds__(128) void gemm_proj_kernel(const u16* __restrict__ Yb,
                                                        const u16* __restrict__ W,
                                                        float* __restrict__ out) {
  const int lane = threadIdx.x & 31;
  const int gw   = blockIdx.x * 4 + (threadIdx.x >> 5);   // 1024 waves
  const int m0   = (gw >> 4) * 64;
  const int n0   = (gw & 15) * 64;
  const int r    = lane & 15;
  const int g    = lane >> 4;

  v8f acc[4][4] = {};
  for (int k0 = 0; k0 < 1024; k0 += 32) {
    v16bf a[4], b[4];
#pragma unroll
    for (int i = 0; i < 4; ++i) a[i] = load_frag_a(Yb, 1024, m0 + i * 16 + r, k0, lane);
#pragma unroll
    for (int j = 0; j < 4; ++j) b[j] = load_frag_b(W, 1024, n0 + j * 16 + r, k0, lane);
    if (k0 + 32 < 1024) {
      __builtin_prefetch(&Yb[(size_t)(m0 + r) * 1024 + k0 + 32], 0, 3);
      __builtin_prefetch(&W[(size_t)(n0 + r) * 1024 + k0 + 32], 0, 3);
    }
#pragma unroll
    for (int i = 0; i < 4; ++i)
#pragma unroll
      for (int j = 0; j < 4; ++j)
        acc[i][j] = WMMA_BF16(a[i], b[j], acc[i][j]);
  }

#pragma unroll
  for (int i = 0; i < 4; ++i) {
#pragma unroll
    for (int j = 0; j < 4; ++j) {
      const int n  = n0 + j * 16 + r;
      const int mb = m0 + i * 16 + 8 * g;
#pragma unroll
      for (int v = 0; v < 8; ++v)
        out[(size_t)(mb + v) * 1024 + n] = acc[i][j][v];
    }
  }
}

// ---------------------------------------------------------------------------
// Launch
// ---------------------------------------------------------------------------
extern "C" void kernel_launch(void* const* d_in, const int* in_sizes, int n_in,
                              void* d_out, int out_size, void* d_ws, size_t ws_size,
                              hipStream_t stream) {
  const float* x      = (const float*)d_in[0];   // [2,2048,1024]
  const float* w_attn = (const float*)d_in[1];   // [3072,1024]
  const float* w_proj = (const float*)d_in[2];   // [1024,1024]
  float* out = (float*)d_out;                    // [2,2048,1024]

  char* ws = (char*)d_ws;
  u16* xb  = (u16*)ws; ws += (size_t)4096 * 1024 * 2;    // x bf16
  u16* wab = (u16*)ws; ws += (size_t)3072 * 1024 * 2;    // w_attn bf16
  u16* wpb = (u16*)ws; ws += (size_t)1024 * 1024 * 2;    // w_proj bf16
  u16* Qb  = (u16*)ws; ws += (size_t)32 * 2048 * 64 * 2; // Q [B*H,T,D] (pre-scaled)
  u16* Kb  = (u16*)ws; ws += (size_t)32 * 2048 * 64 * 2; // K [B*H,T,D]
  u16* Vt  = (u16*)ws; ws += (size_t)32 * 2048 * 64 * 2; // V^T [B*H,D,T]
  u16* Yb  = (u16*)ws; ws += (size_t)4096 * 1024 * 2;    // attention output bf16

  cvt_f32_bf16<<<(4096 * 1024 + 255) / 256, 256, 0, stream>>>(x, xb, 4096 * 1024);
  cvt_f32_bf16<<<(3072 * 1024 + 255) / 256, 256, 0, stream>>>(w_attn, wab, 3072 * 1024);
  cvt_f32_bf16<<<(1024 * 1024 + 255) / 256, 256, 0, stream>>>(w_proj, wpb, 1024 * 1024);

  gemm_qkv_kernel<<<768, 128, 0, stream>>>(xb, wab, Qb, Kb, Vt);   // 3072 waves
  flash_attn_kernel<<<1024, 128, 0, stream>>>(Qb, Kb, Vt, Yb);     // 1024 blocks x 4 waves
  gemm_proj_kernel<<<256, 128, 0, stream>>>(Yb, wpb, out);         // 1024 waves
}
